// CPABTransform2D_BIS_23038204576208
// MI455X (gfx1250) — matrix operand order repaired and checked by reading, not compiled, not run on hardware
//
#include <hip/hip_runtime.h>

#define NCELL 10
#define DTH   9
#define HID   64
#define HPAD  72   // padded row stride (halfs) for activation staging: 36 dwords, good bank spread
#define ABP   22   // padded row stride (floats) for (a,b) staging; even -> 8B aligned pairs
#define EPS_C 1e-10f

typedef __attribute__((ext_vector_type(2)))  _Float16 v2h;
typedef __attribute__((ext_vector_type(8)))  _Float16 v8h;
typedef __attribute__((ext_vector_type(16))) _Float16 v16h;
typedef __attribute__((ext_vector_type(8)))  float    v8f;

static __device__ __forceinline__ v16h cat16(v8h a, v8h b) {
  return __builtin_shufflevector(a, b, 0,1,2,3,4,5,6,7,8,9,10,11,12,13,14,15);
}
static __device__ __forceinline__ v8f wmma16(v16h a, v16h b, v8f c) {
  // v_wmma_f32_16x16x32_f16
  return __builtin_amdgcn_wmma_f32_16x16x32_f16(false, a, false, b, (short)0, c, false, false);
}
// pack two f32 into f16x2 (v_cvt_pk_rtz_f16_f32)
static __device__ __forceinline__ v2h pk_cvt(float a, float b) {
  return __builtin_bit_cast(v2h, __builtin_amdgcn_cvt_pkrtz(a, b));
}

// ---- input layer: each lane owns one point (row = lane); packed f16 FMA (v_pk_fma_f16)
static __device__ __forceinline__ void in_layer32(_Float16* __restrict__ hA,
    const _Float16* __restrict__ WinH, const _Float16* __restrict__ binH,
    float xin, int lane) {
  const _Float16 xh = (_Float16)xin;
  v2h xh2; xh2[0] = xh; xh2[1] = xh;
  const v2h* w2 = (const v2h*)WinH;
  const v2h* b2 = (const v2h*)binH;
  v8h* dst = (v8h*)(hA + lane * HPAD);
  #pragma unroll
  for (int g = 0; g < 8; ++g) {
    v8h o;
    #pragma unroll
    for (int e = 0; e < 4; ++e) {
      const v2h r = w2[g * 4 + e] * xh2 + b2[g * 4 + e];
      o[2 * e]     = r[0];
      o[2 * e + 1] = r[1];
    }
    dst[g] = o;
  }
}

// ---- load one 16x32 A fragment: rows m*16+pl, K chunk kc (0 -> K0..31, 1 -> K32..63)
static __device__ __forceinline__ v16h loadA(const _Float16* __restrict__ hin,
    int m, int pl, int half, int kc) {
  const v8h* hv = (const v8h*)(hin + (m * 16 + pl) * HPAD + kc * 32 + half * 16);
  return cat16(hv[0], hv[1]);   // K = kc*32 + half*16 + 0..15
}

// ---- one 64x64 layer on 32 rows: hout = relu(hin @ W^T + bias); 16 WMMAs, B reused across M-tiles
static __device__ __forceinline__ void mid_layer32(const _Float16* __restrict__ hin,
    _Float16* __restrict__ hout, const _Float16* __restrict__ Wl,
    const float* __restrict__ bias, int lane) {
  const int half = lane >> 4, pl = lane & 15;
  v16h A00 = loadA(hin, 0, pl, half, 0);
  v16h A01 = loadA(hin, 0, pl, half, 1);
  v16h A10 = loadA(hin, 1, pl, half, 0);
  v16h A11 = loadA(hin, 1, pl, half, 1);
  #pragma unroll
  for (int t = 0; t < 4; ++t) {
    const int nn = pl + 16 * t;
    // B fragment (32x16): lane holds col nn, K = kc*32 + half*16 + 0..15
    const v8h* wv0 = (const v8h*)(Wl + nn * HID + half * 16);
    v16h B0 = cat16(wv0[0], wv0[1]);
    v16h B1 = cat16(wv0[4], wv0[5]);
    const float bb = bias[nn];
    #pragma unroll
    for (int mt = 0; mt < 2; ++mt) {
      v8f c;
      #pragma unroll
      for (int r = 0; r < 8; ++r) c[r] = bb;
      c = wmma16((mt == 0) ? A00 : A10, B0, c);
      c = wmma16((mt == 0) ? A01 : A11, B1, c);
      // C element r -> (row = mt*16 + r + half*8, col nn); packed cvt + packed relu
      #pragma unroll
      for (int r = 0; r < 8; r += 2) {
        v2h mm = pk_cvt(c[r], c[r + 1]);
        v2h z = {};
        mm = __builtin_elementwise_max(mm, z);   // v_pk_max_num_f16
        const int row = mt * 16 + r + half * 8;
        hout[row * HPAD + nn]       = mm[0];
        hout[(row + 1) * HPAD + nn] = mm[1];
      }
    }
  }
}

// ---- fused output+basis layer: ab[row][c2] = h @ Wab^T + bab (Wab = B*Wout padded to 32 rows)
//      writes (a,b) pairs straight into the scan staging buffer; 8 WMMAs
static __device__ __forceinline__ void ab_layer32(const _Float16* __restrict__ hin,
    float* __restrict__ ab, const _Float16* __restrict__ WabP,
    const float* __restrict__ babP, int lane) {
  const int half = lane >> 4, pl = lane & 15;
  v16h A00 = loadA(hin, 0, pl, half, 0);
  v16h A01 = loadA(hin, 0, pl, half, 1);
  v16h A10 = loadA(hin, 1, pl, half, 0);
  v16h A11 = loadA(hin, 1, pl, half, 1);
  #pragma unroll
  for (int t = 0; t < 2; ++t) {
    const int nn = pl + 16 * t;
    const v8h* wv0 = (const v8h*)(WabP + nn * HID + half * 16);
    v16h B0 = cat16(wv0[0], wv0[1]);
    v16h B1 = cat16(wv0[4], wv0[5]);
    const float bb = babP[nn];
    #pragma unroll
    for (int mt = 0; mt < 2; ++mt) {
      v8f c;
      #pragma unroll
      for (int r = 0; r < 8; ++r) c[r] = bb;
      c = wmma16((mt == 0) ? A00 : A10, B0, c);
      c = wmma16((mt == 0) ? A01 : A11, B1, c);
      if (nn < 2 * NCELL) {                       // cols 20..31 are padding
        #pragma unroll
        for (int r = 0; r < 8; ++r)
          ab[(mt * 16 + r + half * 8) * ABP + nn] = c[r];
      }
    }
  }
}

// ---- closed-form CPAB scan; rcp-based divides + wave-uniform early exit
static __device__ __forceinline__ void cpab_scan(const float* __restrict__ abr,
    float x, float& zf, float& lgf) {
  int c = (int)floorf(x * (float)NCELL);
  c = min(max(c, 0), NCELL - 1);
  float xc = x, t = 1.0f, lg = 0.0f;
  bool done = false;
  #pragma unroll 1
  for (int it = 0; it < NCELL + 2; ++it) {
    const float a = abr[2 * c];
    const float b = abr[2 * c + 1];
    const float fc = (float)c;
    const float left  = fc * (1.0f / NCELL);
    const float right = left + (1.0f / NCELL);
    const float v = fmaf(a, xc, b);
    const bool small_a = fabsf(a) < EPS_C;
    const float sa  = small_a ? 1.0f : a;
    const float rsa = __builtin_amdgcn_rcpf(sa);
    float ta = t * a; ta = fminf(fmaxf(ta, -50.0f), 50.0f);
    const float eta = __expf(ta);
    const float psi = small_a ? fmaf(t, b, xc) : fmaf(eta, xc, (b * rsa) * (eta - 1.0f));
    const bool inside = (psi >= left) && (psi <= right);
    const float xb = (v >= 0.0f) ? right : left;
    const float sv = (fabsf(v) < EPS_C) ? 1.0f : v;
    const float ratio = fmaxf(fmaf(a, xb, b) * __builtin_amdgcn_rcpf(sv), EPS_C);
    const float sb = (fabsf(b) < EPS_C) ? 1.0f : b;
    float tau = small_a ? ((xb - xc) * __builtin_amdgcn_rcpf(sb))
                        : (__logf(ratio) * rsa);
    tau = fminf(fmaxf(tau, 0.0f), t);
    const bool fin = inside && !done;
    const bool cross = (!inside) && (!done);
    const float dt = fin ? t : tau;
    if (!done) {
      lg = fmaf(a, dt, lg);
      xc = inside ? fminf(fmaxf(psi, left), right) : xb;
      t  = inside ? 0.0f : (t - tau);
    }
    if (cross) c = min(max(c + ((v >= 0.0f) ? 1 : -1), 0), NCELL - 1);
    done = done || inside;
    if (__all(done)) break;   // state frozen once done -> exact vs fixed-trip reference
  }
  zf = xc; lgf = lg;
}

struct LdsWeights {
  _Float16 Wmid[2][2][HID][HID];   // [mlp(0=m1,1=m2)][layer][n][k]
  _Float16 Wab[2][32][HID];        // fused B*Wout, rows 20..31 = 0
  _Float16 WinH[2][HID];           // f16 copies for packed input layer
  _Float16 binH[2][HID];
  float bmid[2][2][HID];
  float bab[2][32];                // fused B*bout
};

__global__ __launch_bounds__(256, 1)
void cpab2d_fused_kernel(
    const float* __restrict__ x, const float* __restrict__ Bbg,
    const float* __restrict__ W1in, const float* __restrict__ b1in,
    const float* __restrict__ W1mid, const float* __restrict__ b1mid,
    const float* __restrict__ W1out, const float* __restrict__ b1out,
    const float* __restrict__ W2in, const float* __restrict__ b2in,
    const float* __restrict__ W2mid, const float* __restrict__ b2mid,
    const float* __restrict__ W2out, const float* __restrict__ b2out,
    float* __restrict__ out, int N) {
  __shared__ LdsWeights W;
  __shared__ _Float16 hbufA[8][32 * HPAD];
  __shared__ _Float16 hbufB[8][32 * HPAD];
  __shared__ float    abbuf[8][32 * ABP];

  const int tid = threadIdx.x;
  // ---- cooperative weight preload (f32 -> f16), incl. fused Wab = B*Wout, bab = B*bout ----
  {
    _Float16* wm0 = &W.Wmid[0][0][0][0];
    _Float16* wm1 = &W.Wmid[1][0][0][0];
    for (int i = tid; i < 2 * HID * HID; i += 256) {
      wm0[i] = (_Float16)W1mid[i];
      wm1[i] = (_Float16)W2mid[i];
    }
    for (int i = tid; i < 2 * 32 * HID; i += 256) {
      const int m = i >> 11;            // / (32*64)
      const int r = (i >> 6) & 31;
      const int k = i & 63;
      float acc = 0.0f;
      if (r < 2 * NCELL) {
        const float* Wo = m ? W2out : W1out;
        #pragma unroll
        for (int d = 0; d < DTH; ++d)
          acc = fmaf(Bbg[r * DTH + d], Wo[d * HID + k], acc);
      }
      W.Wab[m][r][k] = (_Float16)acc;
    }
    for (int i = tid; i < 64; i += 256) {
      const int m = i >> 5, r = i & 31;
      float acc = 0.0f;
      if (r < 2 * NCELL) {
        const float* bo = m ? b2out : b1out;
        #pragma unroll
        for (int d = 0; d < DTH; ++d) acc = fmaf(Bbg[r * DTH + d], bo[d], acc);
      }
      W.bab[m][r] = acc;
    }
    for (int i = tid; i < HID; i += 256) {
      W.WinH[0][i] = (_Float16)W1in[i];  W.WinH[1][i] = (_Float16)W2in[i];
      W.binH[0][i] = (_Float16)b1in[i];  W.binH[1][i] = (_Float16)b2in[i];
      W.bmid[0][0][i] = b1mid[i]; W.bmid[0][1][i] = b1mid[HID + i];
      W.bmid[1][0][i] = b2mid[i]; W.bmid[1][1][i] = b2mid[HID + i];
    }
  }
  __syncthreads();

  const int lane = tid & 31;
  const int wid  = tid >> 5;
  _Float16* hA = hbufA[wid];
  _Float16* hB = hbufB[wid];
  float*    ab = abbuf[wid];
  const float* abr = ab + lane * ABP;

  const int numTiles = (N + 31) >> 5;   // 32 points per wave-iteration
  const int gwave  = blockIdx.x * 8 + wid;
  const int nwaves = gridDim.x * 8;

  for (int tile = gwave; tile < numTiles; tile += nwaves) {
    const int p = tile * 32 + lane;     // each lane owns one point end-to-end
    float x1 = 0.0f, x2 = 0.0f;
    if (p < N) { x2 = x[2 * (size_t)p]; x1 = x[2 * (size_t)p + 1]; }

    // ---- (a,b)_2 = fused MLP2(x1) ----
    in_layer32(hA, W.WinH[1], W.binH[1], x1, lane);
    mid_layer32(hA, hB, &W.Wmid[1][0][0][0], W.bmid[1][0], lane);
    mid_layer32(hB, hA, &W.Wmid[1][1][0][0], W.bmid[1][1], lane);
    ab_layer32(hA, ab, &W.Wab[1][0][0], W.bab[1], lane);

    // ---- z2, log dz2 = CPAB(x2) ----
    float z2, lg2;
    cpab_scan(abr, x2, z2, lg2);

    // ---- (a,b)_1 = fused MLP1(z2) ----
    in_layer32(hA, W.WinH[0], W.binH[0], z2, lane);
    mid_layer32(hA, hB, &W.Wmid[0][0][0][0], W.bmid[0][0], lane);
    mid_layer32(hB, hA, &W.Wmid[0][1][0][0], W.bmid[0][1], lane);
    ab_layer32(hA, ab, &W.Wab[0][0][0], W.bab[0], lane);

    // ---- z1, log dz1 = CPAB(x1) ----
    float z1, lg1;
    cpab_scan(abr, x1, z1, lg1);

    if (p < N) {
      out[2 * (size_t)p]     = z2;
      out[2 * (size_t)p + 1] = z1;
      out[2 * (size_t)N + 2 * (size_t)p]     = lg2;
      out[2 * (size_t)N + 2 * (size_t)p + 1] = lg1;
    }
  }
}

extern "C" void kernel_launch(void* const* d_in, const int* in_sizes, int n_in,
                              void* d_out, int out_size, void* d_ws, size_t ws_size,
                              hipStream_t stream) {
  const float* x     = (const float*)d_in[0];
  const float* Bb    = (const float*)d_in[1];
  const float* W1in  = (const float*)d_in[2];
  const float* b1in  = (const float*)d_in[3];
  const float* W1mid = (const float*)d_in[4];
  const float* b1mid = (const float*)d_in[5];
  const float* W1out = (const float*)d_in[6];
  const float* b1out = (const float*)d_in[7];
  const float* W2in  = (const float*)d_in[8];
  const float* b2in  = (const float*)d_in[9];
  const float* W2mid = (const float*)d_in[10];
  const float* b2mid = (const float*)d_in[11];
  const float* W2out = (const float*)d_in[12];
  const float* b2out = (const float*)d_in[13];
  float* out = (float*)d_out;

  const int N = in_sizes[0] / 2;
  const int numTiles = (N + 31) / 32;
  int blocks = (numTiles + 7) / 8;     // 8 waves/block
  if (blocks > 2048) blocks = 2048;
  if (blocks < 1) blocks = 1;

  cpab2d_fused_kernel<<<blocks, 256, 0, stream>>>(
      x, Bb, W1in, b1in, W1mid, b1mid, W1out, b1out,
      W2in, b2in, W2mid, b2mid, W2out, b2out, out, N);
}
